// QuantizedLinear_38766374814111
// MI455X (gfx1250) — matrix operand order repaired and checked by reading, not compiled, not run on hardware
//
#include <hip/hip_runtime.h>
#include <cstddef>

// ---------------------------------------------------------------------------
// Quantized linear  out[b,s,o] = sum_k x[b,s,k] * (w[o,k]*scale[o]) + bias[o]
// as a tiled fp16-WMMA GEMM: M=B*S=4096, N=OUT=16384, K=IN=4096.
//
// Weights are int8-valued -> exactly representable in fp16, so the WMMA B
// operand carries the raw integer value in fp16 (exact); scale/bias are
// applied in the f32 epilogue. Only x suffers f32->f16 rounding.
// ---------------------------------------------------------------------------

typedef _Float16 h4  __attribute__((ext_vector_type(4)));
typedef _Float16 h8  __attribute__((ext_vector_type(8)));
typedef _Float16 h16 __attribute__((ext_vector_type(16)));
typedef float    f4  __attribute__((ext_vector_type(4)));
typedef float    v8f __attribute__((ext_vector_type(8)));
typedef int      i4  __attribute__((ext_vector_type(4)));

#define TM   128   // block M tile
#define TN   128   // block N tile
#define TK   32    // K step = one v_wmma_f32_16x16x32_f16
#define LDSH 40    // LDS row stride in halves (80 B: 16B-aligned, conflict-free)

__global__ __launch_bounds__(256) void qlinear_wmma_kernel(
    const float* __restrict__ x,      // [M,K] f32
    const int*   __restrict__ w,      // [N,K] int8 values in int32
    const float* __restrict__ scale,  // [N]
    const float* __restrict__ bias,   // [N]
    float*       __restrict__ out,    // [M,N]
    int M, int N, int K)
{
    __shared__ _Float16 Ash[2][TM * LDSH];  // x tile, fp16
    __shared__ _Float16 Bsh[2][TN * LDSH];  // w tile, exact int-in-fp16

    const int tid   = threadIdx.x;
    const int lane  = tid & 31;
    const int wave  = tid >> 5;      // 8 waves
    const int waveM = wave & 3;      // 4 waves along M -> 32 rows each
    const int waveN = wave >> 2;     // 2 waves along N -> 64 cols each
    const int mBase = blockIdx.y * TM;
    const int nBase = blockIdx.x * TN;
    const int r16   = lane & 15;

    // global->reg staging (4 x 16B vectors each for A and B per thread)
    f4 aReg[4];
    i4 bReg[4];

    auto loadGlobal = [&](int it) {
        const int kBase = it * TK;
        #pragma unroll
        for (int j = 0; j < 4; ++j) {
            int idx = tid + 256 * j;          // 0..1023 vec4 slots
            int row = idx >> 3;               // 128 rows, 8 vec4 per row
            int kq  = (idx & 7) << 2;         // k offset within tile
            aReg[j] = *(const f4*)(x + (size_t)(mBase + row) * K + kBase + kq);
            bReg[j] = *(const i4*)(w + (size_t)(nBase + row) * K + kBase + kq);
        }
    };

    auto stageLDS = [&](int buf) {
        #pragma unroll
        for (int j = 0; j < 4; ++j) {
            int idx = tid + 256 * j;
            int row = idx >> 3;
            int kq  = (idx & 7) << 2;
            h4 av = { (_Float16)aReg[j].x, (_Float16)aReg[j].y,
                      (_Float16)aReg[j].z, (_Float16)aReg[j].w };
            // int values in [-127,127]: exact via i16 convert
            h4 bv = { (_Float16)(short)bReg[j].x, (_Float16)(short)bReg[j].y,
                      (_Float16)(short)bReg[j].z, (_Float16)(short)bReg[j].w };
            *(h4*)&Ash[buf][row * LDSH + kq] = av;
            *(h4*)&Bsh[buf][row * LDSH + kq] = bv;
        }
    };

    v8f acc[2][4];
    #pragma unroll
    for (int i = 0; i < 2; ++i)
        #pragma unroll
        for (int j = 0; j < 4; ++j)
            acc[i][j] = (v8f){0.f,0.f,0.f,0.f,0.f,0.f,0.f,0.f};

    const int KT = K / TK;

    loadGlobal(0);
    stageLDS(0);
    __syncthreads();

    for (int it = 0; it < KT; ++it) {
        const int  cur  = it & 1;
        const bool more = (it + 1) < KT;
        if (more) loadGlobal(it + 1);   // prefetch next tile while computing

        const _Float16* As = Ash[cur];
        const _Float16* Bs = Bsh[cur];
        // ISA 7.12.2 wave32 layouts:
        //  A (16x32 f16): lanes 0-15 K=[0..7],[16..23]; lanes 16-31 K=[8..15],[24..31]
        //  B (32x16 f16): lanes 0-15 K=[0..7],[16..23]; lanes 16-31 K=[8..15],[24..31]
        const int kloA = (lane < 16) ? 0 : 8;
        const int kloB = (lane < 16) ? 0 : 8;

        h16 a[2], b[4];
        #pragma unroll
        for (int mt = 0; mt < 2; ++mt) {
            const _Float16* p = As + (waveM * 32 + mt * 16 + r16) * LDSH;
            h8 lo = *(const h8*)(p + kloA);
            h8 hi = *(const h8*)(p + kloA + 16);
            a[mt] = __builtin_shufflevector(lo, hi,
                        0,1,2,3,4,5,6,7,8,9,10,11,12,13,14,15);
        }
        #pragma unroll
        for (int nt = 0; nt < 4; ++nt) {
            const _Float16* p = Bs + (waveN * 64 + nt * 16 + r16) * LDSH + kloB;
            h8 lo = *(const h8*)(p);
            h8 hi = *(const h8*)(p + 16);
            b[nt] = __builtin_shufflevector(lo, hi,
                        0,1,2,3,4,5,6,7,8,9,10,11,12,13,14,15);
        }

        #pragma unroll
        for (int mt = 0; mt < 2; ++mt)
            #pragma unroll
            for (int nt = 0; nt < 4; ++nt)
                acc[mt][nt] = __builtin_amdgcn_wmma_f32_16x16x32_f16(
                    /*neg_a=*/false, a[mt], /*neg_b=*/false, b[nt],
                    /*c_mod=*/(short)0, acc[mt][nt],
                    /*reuse_a=*/false, /*reuse_b=*/false);

        __syncthreads();                 // everyone done reading buffer `cur`
        if (more) stageLDS(cur ^ 1);     // fill the other buffer
        __syncthreads();                 // writes visible before next compute
    }

    // Epilogue: C/D layout -> VGPR v holds M = v (+8 for lanes 16-31), N = lane%16.
    // out = acc * scale[n] + bias[n] in f32.
    #pragma unroll
    for (int mt = 0; mt < 2; ++mt) {
        #pragma unroll
        for (int nt = 0; nt < 4; ++nt) {
            const int   n  = nBase + waveN * 64 + nt * 16 + r16;
            const float s  = scale[n];
            const float bz = bias[n];
            const int   m0 = mBase + waveM * 32 + mt * 16 + ((lane < 16) ? 0 : 8);
            float* po = out + (size_t)m0 * N + n;
            #pragma unroll
            for (int v = 0; v < 8; ++v)
                po[(size_t)v * N] = acc[mt][nt][v] * s + bz;
        }
    }
}

extern "C" void kernel_launch(void* const* d_in, const int* in_sizes, int n_in,
                              void* d_out, int out_size, void* d_ws, size_t ws_size,
                              hipStream_t stream) {
    (void)n_in; (void)out_size; (void)d_ws; (void)ws_size;
    const float* x     = (const float*)d_in[0];
    const int*   w     = (const int*)  d_in[1];
    const float* scale = (const float*)d_in[2];
    const float* bias  = (const float*)d_in[3];
    float*       out   = (float*)d_out;

    const int N = in_sizes[3];           // OUT = 16384
    const int K = in_sizes[1] / N;       // IN  = 4096
    const int M = in_sizes[0] / K;       // B*S = 4096

    dim3 grid(N / TN, M / TM);           // (128, 32)
    qlinear_wmma_kernel<<<grid, dim3(256), 0, stream>>>(x, w, scale, bias, out, M, N, K);
}
